// KeypointGraph_23502061044365
// MI455X (gfx1250) — matrix operand order repaired and checked by reading, not compiled, not run on hardware
//
#include <hip/hip_runtime.h>

// ---------------------------------------------------------------------------
// 2-layer GAT for MI455X (gfx1250, wave32).
// GEMMs run on the matrix pipes via v_wmma_f32_16x16x32_bf16 (fp32->bf16
// inputs, f32 accumulate). Edge softmax/scatter uses f32 global atomics,
// one wave32 per (edge, head) for coalesced 256-wide message rows.
// Fixed problem shape: F=H=256, HEADS=4, N = B*K (8704), divisible by 32.
// ---------------------------------------------------------------------------

typedef __attribute__((ext_vector_type(16))) __bf16 v16bf;
typedef __attribute__((ext_vector_type(8)))  __bf16 v8bf;
typedef __attribute__((ext_vector_type(8)))  float  v8f;

#define FDIM   256
#define NOUT   1024      // HEADS * C
#define HEADS  4
#define NEG_SLOPE 0.2f

// ---------------------------------------------------------------- converts
__global__ __launch_bounds__(256) void k_cvt_x(const float* __restrict__ x,
                                               __bf16* __restrict__ xb, int total) {
  int i = blockIdx.x * 256 + threadIdx.x;
  if (i < total) xb[i] = (__bf16)x[i];
}

// W [256 x 1024] fp32 row-major  ->  WT [1024 x 256] bf16 row-major
__global__ __launch_bounds__(256) void k_cvt_wt(const float* __restrict__ W,
                                                __bf16* __restrict__ WT) {
  int i  = blockIdx.x * 256 + threadIdx.x;   // 1024*256 threads
  int no = i >> 8;            // output column (0..1023)
  int k  = i & 255;           // reduction index
  WT[no * FDIM + k] = (__bf16)W[k * NOUT + no];
}

// ---------------------------------------------------------------- WMMA GEMM
// C[N x 1024] = A[N x 256] (bf16) * B[256 x 1024] (bf16, given as B^T).
// Block = 8 waves, tile 32 rows x 64 cols; each wave owns one 16x16 tile.
// K loop: 8 steps of K=32, fully unrolled -> 8 v_wmma per wave.
// Fragment layouts follow CDNA5 ISA 7.12.2:
//   A 16x32 bf16: lane(0-15)=row M, khalf=lane>>4; elems 0-7 -> K=khalf*8+j,
//                 elems 8-15 -> K=16+khalf*8+j  (two contiguous 16B loads)
//   B 32x16 bf16: col=lane&15; same K order as A: elems 0-7 -> K=khalf*8+j,
//                 elems 8-15 -> K=16+khalf*8+j  (two 16B loads from B^T row).
__global__ __launch_bounds__(256) void k_gemm_wmma(const __bf16* __restrict__ A,
                                                   const __bf16* __restrict__ BT,
                                                   float* __restrict__ C) {
  const int waveId = threadIdx.x >> 5;
  const int lane   = threadIdx.x & 31;
  const int m0     = blockIdx.x * 32 + (waveId >> 2) * 16;
  const int n0     = blockIdx.y * 64 + (waveId & 3) * 16;
  const int row    = m0 + (lane & 15);
  const int col    = n0 + (lane & 15);
  const int khalf  = lane >> 4;

  v8f acc = {};
#pragma unroll
  for (int kk = 0; kk < FDIM; kk += 32) {
    v8bf alo = *(const v8bf*)(A + row * FDIM + kk + khalf * 8);
    v8bf ahi = *(const v8bf*)(A + row * FDIM + kk + 16 + khalf * 8);
    v16bf a, b;
#pragma unroll
    for (int j = 0; j < 8; ++j) { a[j] = alo[j]; a[8 + j] = ahi[j]; }
    v8bf blo = *(const v8bf*)(BT + col * FDIM + kk + khalf * 8);
    v8bf bhi = *(const v8bf*)(BT + col * FDIM + kk + 16 + khalf * 8);
#pragma unroll
    for (int j = 0; j < 8; ++j) { b[j] = blo[j]; b[8 + j] = bhi[j]; }
    acc = __builtin_amdgcn_wmma_f32_16x16x32_bf16(
        /*neg_a=*/false, a, /*neg_b=*/false, b,
        /*c_mod=*/(short)0, acc, /*reuse_a=*/false, /*reuse_b=*/false);
  }
  // C/D layout: VGPR r -> M = m0 + khalf*8 + r, N = n0 + (lane&15)
  const int crow = m0 + khalf * 8;
#pragma unroll
  for (int r = 0; r < 8; ++r) C[(crow + r) * NOUT + col] = acc[r];
}

// ----------------------------------------------------- attention coefficients
// one wave per (node, head): alpha_src/dst = dot(h[n,h,:], a_vec[h,:])
__global__ __launch_bounds__(256) void k_alpha(const float* __restrict__ h,
                                               const float* __restrict__ a_src,
                                               const float* __restrict__ a_dst,
                                               float* __restrict__ asrcN,
                                               float* __restrict__ adstN, int nUnits) {
  int unit = (blockIdx.x * 256 + threadIdx.x) >> 5;
  int lane = threadIdx.x & 31;
  if (unit >= nUnits) return;
  int hd = unit & 3, node = unit >> 2;
  const float* hp = h + node * NOUT + hd * FDIM;
  float s1 = 0.f, s2 = 0.f;
#pragma unroll
  for (int c = lane; c < FDIM; c += 32) {
    float v = hp[c];
    s1 += v * a_src[hd * FDIM + c];
    s2 += v * a_dst[hd * FDIM + c];
  }
#pragma unroll
  for (int off = 16; off; off >>= 1) {
    s1 += __shfl_xor(s1, off, 32);
    s2 += __shfl_xor(s2, off, 32);
  }
  if (lane == 0) { asrcN[unit] = s1; adstN[unit] = s2; }
}

// ---------------------------------------------------------------- init state
__global__ __launch_bounds__(256) void k_init(float* __restrict__ accum,
                                              float* __restrict__ emax,
                                              float* __restrict__ denom, int n, int nh) {
  int i = blockIdx.x * 256 + threadIdx.x;
  if (i < n) accum[i] = 0.f;
  if (i < nh) { emax[i] = -1e30f; denom[i] = 0.f; }
}

__device__ __forceinline__ void edge_pair(const long long* __restrict__ ei,
                                          int E, int e, int& s, int& d) {
  if (e < E) { s = (int)ei[e]; d = (int)ei[E + e]; }
  else       { s = d = e - E; }                 // appended self-loops
}

__device__ __forceinline__ float leaky(float v) {
  return v > 0.f ? v : NEG_SLOPE * v;
}

__device__ __forceinline__ void atomicMaxF(float* addr, float val) {
  int* ai = (int*)addr;
  int old = __float_as_int(*addr);
  while (__int_as_float(old) < val) {
    int assumed = old;
    old = atomicCAS(ai, assumed, __float_as_int(val));
    if (old == assumed) break;
  }
}

// ------------------------------------------------------------- edge softmax
__global__ __launch_bounds__(256) void k_edge_max(const long long* __restrict__ ei, int E,
                                                  int Etot,
                                                  const float* __restrict__ asrcN,
                                                  const float* __restrict__ adstN,
                                                  float* __restrict__ emax) {
  int e = blockIdx.x * 256 + threadIdx.x;
  if (e >= Etot) return;
  int s, d; edge_pair(ei, E, e, s, d);
#pragma unroll
  for (int h = 0; h < HEADS; ++h)
    atomicMaxF(&emax[d * HEADS + h], leaky(asrcN[s * HEADS + h] + adstN[d * HEADS + h]));
}

__global__ __launch_bounds__(256) void k_edge_exp(const long long* __restrict__ ei, int E,
                                                  int Etot,
                                                  const float* __restrict__ asrcN,
                                                  const float* __restrict__ adstN,
                                                  const float* __restrict__ emax,
                                                  float* __restrict__ exbuf,
                                                  float* __restrict__ denom) {
  int e = blockIdx.x * 256 + threadIdx.x;
  if (e >= Etot) return;
  int s, d; edge_pair(ei, E, e, s, d);
#pragma unroll
  for (int h = 0; h < HEADS; ++h) {
    float v = leaky(asrcN[s * HEADS + h] + adstN[d * HEADS + h]);
    float ex = __expf(v - emax[d * HEADS + h]);
    exbuf[e * HEADS + h] = ex;
    atomicAdd(&denom[d * HEADS + h], ex);
  }
}

// one wave per (edge, head): accum[dst,h,:] += alpha * h[src,h,:]
__global__ __launch_bounds__(256) void k_scatter(const long long* __restrict__ ei, int E,
                                                 int nUnits,
                                                 const float* __restrict__ exbuf,
                                                 const float* __restrict__ denom,
                                                 const float* __restrict__ hbuf,
                                                 float* __restrict__ accum) {
  int unit = (blockIdx.x * 256 + threadIdx.x) >> 5;
  int lane = threadIdx.x & 31;
  if (unit >= nUnits) return;
  int h = unit & 3, e = unit >> 2;
  int s, d; edge_pair(ei, E, e, s, d);
  float alpha = exbuf[e * HEADS + h] / denom[d * HEADS + h];
  const float* sp = hbuf + s * NOUT + h * FDIM;
  float* dp       = accum + d * NOUT + h * FDIM;
#pragma unroll
  for (int c = lane; c < FDIM; c += 32)
    atomicAdd(&dp[c], alpha * sp[c]);
}

// ------------------------------------------------------ head-mean + bias (+relu)
__global__ __launch_bounds__(256) void k_finalize(const float* __restrict__ accum,
                                                  const float* __restrict__ bias,
                                                  float* __restrict__ outf,
                                                  __bf16* __restrict__ outb,
                                                  int relu_bf16) {
  int i = blockIdx.x * 256 + threadIdx.x;   // N * 256 threads
  int node = i >> 8, c = i & 255;
  float s = 0.f;
#pragma unroll
  for (int h = 0; h < HEADS; ++h) s += accum[node * NOUT + h * FDIM + c];
  s = s * 0.25f + bias[c];
  if (relu_bf16) outb[node * FDIM + c] = (__bf16)fmaxf(s, 0.f);
  else           outf[node * FDIM + c] = s;
}

// ---------------------------------------------------------------------------
extern "C" void kernel_launch(void* const* d_in, const int* in_sizes, int n_in,
                              void* d_out, int out_size, void* d_ws, size_t ws_size,
                              hipStream_t stream) {
  const float*     x   = (const float*)d_in[0];
  const long long* ei  = (const long long*)d_in[1];   // jnp.int64 [2, E]
  const float*     W1  = (const float*)d_in[2];
  const float*     as1 = (const float*)d_in[3];
  const float*     ad1 = (const float*)d_in[4];
  const float*     b1  = (const float*)d_in[5];
  const float*     W2  = (const float*)d_in[6];
  const float*     as2 = (const float*)d_in[7];
  const float*     ad2 = (const float*)d_in[8];
  const float*     b2  = (const float*)d_in[9];
  float* out = (float*)d_out;

  const int N    = in_sizes[0] / FDIM;     // 8704, divisible by 32
  const int E    = in_sizes[1] / 2;        // 139264
  const int Etot = E + N;                  // + self-loops

  // ----- workspace carve-up (256B aligned) -----
  auto align = [](size_t v) { return (v + 255) & ~size_t(255); };
  char* p = (char*)d_ws;
  __bf16* xb    = (__bf16*)p; p += align((size_t)N * FDIM * sizeof(__bf16));
  __bf16* WT    = (__bf16*)p; p += align((size_t)NOUT * FDIM * sizeof(__bf16));
  float*  hbuf  = (float*)p;  p += align((size_t)N * NOUT * sizeof(float));
  float*  accum = (float*)p;  p += align((size_t)N * NOUT * sizeof(float));
  float*  asrcN = (float*)p;  p += align((size_t)N * HEADS * sizeof(float));
  float*  adstN = (float*)p;  p += align((size_t)N * HEADS * sizeof(float));
  float*  emax  = (float*)p;  p += align((size_t)N * HEADS * sizeof(float));
  float*  denom = (float*)p;  p += align((size_t)N * HEADS * sizeof(float));
  float*  exbuf = (float*)p;  p += align((size_t)Etot * HEADS * sizeof(float));
  (void)ws_size; (void)n_in; (void)out_size;

  const int eb   = (Etot + 255) / 256;           // edge-per-thread blocks
  const int ewb  = (Etot * HEADS + 7) / 8;       // (edge,head)-per-wave blocks
  const int nwb  = (N * HEADS + 7) / 8;          // (node,head)-per-wave blocks
  const dim3 gG(N / 32, NOUT / 64);              // WMMA GEMM grid

  // ================= conv1 =================
  k_cvt_x  <<<(N * FDIM + 255) / 256, 256, 0, stream>>>(x, xb, N * FDIM);
  k_cvt_wt <<<(NOUT * FDIM) / 256,    256, 0, stream>>>(W1, WT);
  k_gemm_wmma<<<gG, 256, 0, stream>>>(xb, WT, hbuf);
  k_alpha  <<<nwb, 256, 0, stream>>>(hbuf, as1, ad1, asrcN, adstN, N * HEADS);
  k_init   <<<(N * NOUT + 255) / 256, 256, 0, stream>>>(accum, emax, denom,
                                                        N * NOUT, N * HEADS);
  k_edge_max<<<eb, 256, 0, stream>>>(ei, E, Etot, asrcN, adstN, emax);
  k_edge_exp<<<eb, 256, 0, stream>>>(ei, E, Etot, asrcN, adstN, emax, exbuf, denom);
  k_scatter <<<ewb, 256, 0, stream>>>(ei, E, Etot * HEADS, exbuf, denom, hbuf, accum);
  k_finalize<<<N, 256, 0, stream>>>(accum, b1, nullptr, xb, /*relu+bf16*/1);

  // ================= conv2 =================
  k_cvt_wt <<<(NOUT * FDIM) / 256, 256, 0, stream>>>(W2, WT);
  k_gemm_wmma<<<gG, 256, 0, stream>>>(xb, WT, hbuf);
  k_alpha  <<<nwb, 256, 0, stream>>>(hbuf, as2, ad2, asrcN, adstN, N * HEADS);
  k_init   <<<(N * NOUT + 255) / 256, 256, 0, stream>>>(accum, emax, denom,
                                                        N * NOUT, N * HEADS);
  k_edge_max<<<eb, 256, 0, stream>>>(ei, E, Etot, asrcN, adstN, emax);
  k_edge_exp<<<eb, 256, 0, stream>>>(ei, E, Etot, asrcN, adstN, emax, exbuf, denom);
  k_scatter <<<ewb, 256, 0, stream>>>(ei, E, Etot * HEADS, exbuf, denom, hbuf, accum);
  k_finalize<<<N, 256, 0, stream>>>(accum, b2, out, nullptr, /*fp32 out*/0);
}
